// LSTMCell__43688407335067
// MI455X (gfx1250) — matrix operand order repaired and checked by reading, not compiled, not run on hardware
//
#include <hip/hip_runtime.h>
#include <math.h>

// LSTM cell: B=32768, I=H=512.
// S = [x|h0] (32768x1024) @ [Wi|Wh]^T (2048x1024), then gate math.
// Weights prepacked f32->bf16 (4MB, L2-resident). Main kernel: bf16 WMMA
// GEMM, double-buffered LDS with async global->LDS DMA for weights,
// fused branch-free LSTM epilogue. Output = h1 [B,H] ++ c1 [B,H].

#define B_SZ 32768
#define I_SZ 512
#define H_SZ 512
#define K_SZ 1024
#define N_SZ 2048

#define BM 128   // batch rows per workgroup
#define BN 128   // gate-cols per workgroup (4 gates x 32 h, gate-interleaved)
#define BK 64    // K chunk
#define NCHUNK (K_SZ / BK)
#define LDA 72   // padded LDS stride (bf16 elems)
#define LDB 72

typedef __attribute__((ext_vector_type(16))) __bf16 v16bf;
typedef __attribute__((ext_vector_type(8)))  __bf16 v8bf;
typedef __attribute__((ext_vector_type(2)))  __bf16 v2bf;
typedef __attribute__((ext_vector_type(8)))  float  v8f;
typedef __attribute__((ext_vector_type(4)))  int    v4i;

#define AS1 __attribute__((address_space(1)))
#define AS3 __attribute__((address_space(3)))

#if defined(__has_builtin)
#  if __has_builtin(__builtin_amdgcn_global_load_async_to_lds_b128)
#    define HAS_ASYNC_LDS 1
#  endif
#  if __has_builtin(__builtin_amdgcn_s_wait_asynccnt)
#    define HAS_WAIT_ASYNC 1
#  endif
#  if __has_builtin(__builtin_amdgcn_cvt_pk_bf16_f32)
#    define HAS_CVT_PK_BF16 1
#  endif
#endif

__device__ __forceinline__ unsigned int bf16pk(float a, float b) {
#ifdef HAS_CVT_PK_BF16
  v2bf p = __builtin_amdgcn_cvt_pk_bf16_f32(a, b);
  return *(unsigned int*)&p;
#else
  // single-instruction packed convert (RNE); gfx950+/gfx1250 VOP3
  unsigned int r;
  asm("v_cvt_pk_bf16_f32 %0, %1, %2" : "=v"(r) : "v"(a), "v"(b));
  return r;
#endif
}

__device__ __forceinline__ void async_wait0() {
#ifdef HAS_ASYNC_LDS
#  ifdef HAS_WAIT_ASYNC
  __builtin_amdgcn_s_wait_asynccnt(0);
#  else
  asm volatile("s_wait_asynccnt 0x0" ::: "memory");
#  endif
#endif
}

// branch-free activations (saturate correctly when exp overflows to +inf)
__device__ __forceinline__ float sigmoid_f(float x) {
  return __builtin_amdgcn_rcpf(1.f + __expf(-x));
}
__device__ __forceinline__ float tanh_f(float x) {
  return 1.f - 2.f * __builtin_amdgcn_rcpf(__expf(2.f * x) + 1.f);
}

// ---- prepack: Wc_bf16[2048][1024] (as packed uint pairs) + bias[2048] ----
__global__ void lstm_prepack(const float* __restrict__ Wi,
                             const float* __restrict__ Wh,
                             const float* __restrict__ bi,
                             const float* __restrict__ bh,
                             unsigned int* __restrict__ wc,
                             float* __restrict__ bias) {
  int idx = blockIdx.x * blockDim.x + threadIdx.x;   // one uint = 2 bf16
  if (idx < (N_SZ * K_SZ / 2)) {
    int n  = idx >> 9;        // row 0..2047 (gate*512 + h)
    int k  = (idx & 511) * 2; // k element 0..1022 (even)
    const float* src = (k < I_SZ) ? (Wi + (size_t)n * I_SZ + k)
                                  : (Wh + (size_t)n * H_SZ + (k - I_SZ));
    wc[idx] = bf16pk(src[0], src[1]);
  }
  if (idx < N_SZ) bias[idx] = bi[idx] + bh[idx];
}

// ---- main fused GEMM + LSTM kernel ----
// Column tile j (16 cols) -> gate = j&3, h_sub = j>>2.
// Waves tiled 4(M) x 2(N): wave = 32 rows x 64 cols = one h-slice x 4 gates.
__global__ __launch_bounds__(256)
void lstm_wmma(const float* __restrict__ x,
               const float* __restrict__ h0,
               const float* __restrict__ c0,
               const unsigned int* __restrict__ wc,   // bf16 pairs [2048][512]
               const float* __restrict__ bias,        // [2048]
               float* __restrict__ out) {             // h1 [B,H] ++ c1 [B,H]
  __shared__ unsigned short lA[2][BM * LDA];  // A tile, bf16, [row][k]
  __shared__ unsigned short lB[2][BN * LDB];  // B tile, bf16, [col][k]

  const int tid   = threadIdx.x;
  const int lane  = tid & 31;
  const int wv    = tid >> 5;            // wave 0..7
  const int wm    = wv >> 1;             // 0..3: M sub-tile (32 rows)
  const int wn    = wv & 1;              // 0..1: N group (h_sub)
  const int l16   = lane & 15;
  const int lhi   = lane >> 4;           // 0/1 half-wave
  const int bm0   = blockIdx.x * BM;
  const int hbase = blockIdx.y * 32;

  v8f acc[2][4];                         // [m-tile][gate]
  #pragma unroll
  for (int mi = 0; mi < 2; ++mi)
    #pragma unroll
    for (int jj = 0; jj < 4; ++jj)
      acc[mi][jj] = (v8f){0.f, 0.f, 0.f, 0.f, 0.f, 0.f, 0.f, 0.f};

  // --- staging helpers ---
  auto stageB = [&](int k0, int buf) {   // async DMA: bf16 weights -> LDS
    #pragma unroll
    for (int t = 0; t < 4; ++t) {
      int idx = tid + t * 256;           // 0..1023
      int c   = idx >> 3;                // tile col 0..127
      int ch  = idx & 7;                 // 16B chunk
      int j   = c >> 4;                  // col tile
      int ng  = (j & 3) * H_SZ + hbase + ((j >> 2) << 4) + (c & 15);
      const unsigned int* gsrc = wc + (size_t)ng * (K_SZ / 2) + (k0 >> 1) + ch * 4;
#ifdef HAS_ASYNC_LDS
      __builtin_amdgcn_global_load_async_to_lds_b128(
          (AS1 v4i*)gsrc,
          (AS3 v4i*)&lB[buf][c * LDB + ch * 8],
          /*offset=*/0, /*cpol=*/0);
#else
      *(uint4*)(&lB[buf][c * LDB + ch * 8]) = *(const uint4*)gsrc;
#endif
    }
  };
  auto stageA = [&](int k0, int buf) {   // f32 -> bf16 pack into LDS
    const float* srcA = (k0 < I_SZ) ? (x + k0) : (h0 + (k0 - I_SZ));
    #pragma unroll
    for (int t = 0; t < 8; ++t) {
      int idx = tid + t * 256;           // 0..2047
      int r   = idx >> 4;                // row 0..127
      int seg = idx & 15;                // 4-float segment
      const float4 v = *(const float4*)(srcA + (size_t)(bm0 + r) * I_SZ + seg * 4);
      *(uint2*)(&lA[buf][r * LDA + seg * 4]) =
          make_uint2(bf16pk(v.x, v.y), bf16pk(v.z, v.w));
    }
  };
  auto compute = [&](int buf) {          // 16 WMMA on the current buffers
    #pragma unroll
    for (int ks = 0; ks < 2; ++ks) {
      const int kk = ks * 32;
      const int ka = kk + (lhi << 4);
      v16bf afrag[2];
      #pragma unroll
      for (int mi = 0; mi < 2; ++mi) {
        const int ar = wm * 32 + mi * 16 + l16;
        *(v8bf*)&afrag[mi]       = *(const v8bf*)(&lA[buf][ar * LDA + ka]);
        *((v8bf*)&afrag[mi] + 1) = *(const v8bf*)(&lA[buf][ar * LDA + ka + 8]);
      }
      #pragma unroll
      for (int jj = 0; jj < 4; ++jj) {
        const int bc = ((wn * 4 + jj) << 4) + l16;
        const int kb = kk + (lhi << 4);
        v16bf bfrag;
        *(v8bf*)&bfrag       = *(const v8bf*)(&lB[buf][bc * LDB + kb]);
        *((v8bf*)&bfrag + 1) = *(const v8bf*)(&lB[buf][bc * LDB + kb + 8]);
        #pragma unroll
        for (int mi = 0; mi < 2; ++mi) {
          acc[mi][jj] = __builtin_amdgcn_wmma_f32_16x16x32_bf16(
              false, afrag[mi], false, bfrag,
              (short)0, acc[mi][jj], false, false);
        }
      }
    }
  };

  // --- software-pipelined main loop: stage(c+1) overlaps compute(c) ---
  stageB(0, 0);
  stageA(0, 0);
  async_wait0();
  __syncthreads();
  int buf = 0;
  for (int c = 0; c < NCHUNK; ++c) {
    if (c + 1 < NCHUNK) {
      stageB((c + 1) * BK, buf ^ 1);     // async DMA runs under compute
      stageA((c + 1) * BK, buf ^ 1);     // pack VALU co-executes with WMMA
    }
    compute(buf);
    async_wait0();
    __syncthreads();
    buf ^= 1;
  }

  // --- fused LSTM epilogue (all 4 gates live in the same lane) ---
  const int hg = hbase + (wn << 4) + l16;
  const float b_i = bias[0 * H_SZ + hg];
  const float b_f = bias[1 * H_SZ + hg];
  const float b_g = bias[2 * H_SZ + hg];
  const float b_o = bias[3 * H_SZ + hg];
  #pragma unroll
  for (int mi = 0; mi < 2; ++mi) {
    const int rbase = bm0 + wm * 32 + mi * 16 + (lhi << 3);
    #pragma unroll
    for (int r = 0; r < 8; ++r) {
      const int row = rbase + r;
      const float ig = sigmoid_f(acc[mi][0][r] + b_i);
      const float fg = sigmoid_f(acc[mi][1][r] + b_f);
      const float gg = tanh_f(acc[mi][2][r] + b_g);
      const float og = sigmoid_f(acc[mi][3][r] + b_o);
      const float cp = c0[(size_t)row * H_SZ + hg];
      const float c1 = fg * cp + ig * gg;
      const float h1 = og * tanh_f(c1);
      out[(size_t)row * H_SZ + hg] = h1;
      out[(size_t)B_SZ * H_SZ + (size_t)row * H_SZ + hg] = c1;
    }
  }
}

extern "C" void kernel_launch(void* const* d_in, const int* in_sizes, int n_in,
                              void* d_out, int out_size, void* d_ws, size_t ws_size,
                              hipStream_t stream) {
  const float* x  = (const float*)d_in[0];
  const float* h0 = (const float*)d_in[1];
  const float* c0 = (const float*)d_in[2];
  const float* Wi = (const float*)d_in[3];
  const float* bi = (const float*)d_in[4];
  const float* Wh = (const float*)d_in[5];
  const float* bh = (const float*)d_in[6];
  float* out = (float*)d_out;

  // workspace: 4MB packed bf16 weights + 8KB fused bias
  unsigned int* wc   = (unsigned int*)d_ws;
  float*        bias = (float*)((char*)d_ws + (size_t)N_SZ * K_SZ * 2);

  {
    int total = N_SZ * K_SZ / 2;
    lstm_prepack<<<dim3((total + 255) / 256), dim3(256), 0, stream>>>(
        Wi, Wh, bi, bh, wc, bias);
  }
  lstm_wmma<<<dim3(B_SZ / BM, H_SZ / 32), dim3(256), 0, stream>>>(
      x, h0, c0, wc, bias, out);
}
